// MultiHeadLatentAttention_80083960201772
// MI455X (gfx1250) — matrix operand order repaired and checked by reading, not compiled, not run on hardware
//
#include <hip/hip_runtime.h>
#include <hip/hip_bf16.h>
#include <math.h>

// ---------------------------------------------------------------------------
// MLA forward for MI455X (gfx1250, wave32, WMMA).
// All matmuls: v_wmma_f32_16x16x32_bf16 with hi/lo bf16 split of fp32 operands
// (3 WMMAs: hi*hi + hi*lo + lo*hi, fp32 accumulate) ~= fp32 accuracy on the
// bf16 matrix pipe. Reused operands (weights, roped K, V) are pre-packed ONCE
// into the exact fragment byte layout so hot loops are b128 loads + WMMA.
// Workspace: ~54 MB (fp32 intermediates + packed bf16 fragments).
// ---------------------------------------------------------------------------

typedef __attribute__((ext_vector_type(16))) __bf16 v16bf;
typedef __attribute__((ext_vector_type(8)))  float  v8f;
typedef __attribute__((ext_vector_type(4)))  float  f32x4;   // native clang vector

#define HN      16
#define DH      32
#define LQ      2048
#define DMODEL  2048
#define DKV     512
#define KVLEN   4096

// K index inside a 16-bit A fragment (ISA 7.12.2, 16-bit A 16x32 table)
__device__ __forceinline__ int a_kidx(int j, int half) {
  return (j < 8 ? j : j + 8) + (half ? 8 : 0);
}

// Load the 16 fp32 values of an A-layout fragment row: two contiguous 8-float
// runs -> 4x b128 loads. p must be 16B aligned (rows are 512/2048-float strided).
__device__ __forceinline__ void load16(const float* __restrict__ p, int half,
                                       float v[16]) {
  const f32x4* p4 = (const f32x4*)p;
  f32x4 f0 = p4[half * 2 + 0];   // k = half*8 + 0..3
  f32x4 f1 = p4[half * 2 + 1];   // k = half*8 + 4..7
  f32x4 f2 = p4[half * 2 + 4];   // k = 16 + half*8 + 0..3
  f32x4 f3 = p4[half * 2 + 5];   // k = 16 + half*8 + 4..7
#pragma unroll
  for (int j = 0; j < 4; ++j) {
    v[j]      = f0[j];
    v[4 + j]  = f1[j];
    v[8 + j]  = f2[j];
    v[12 + j] = f3[j];
  }
}

__device__ __forceinline__ void split16(const float v[16], float scale,
                                        v16bf& hi, v16bf& lo) {
#pragma unroll
  for (int j = 0; j < 16; ++j) {
    float x = v[j] * scale;
    __bf16 h = (__bf16)x;
    hi[j] = h;
    lo[j] = (__bf16)(x - (float)h);
  }
}

// Split-bf16 fp32-accurate MAC: acc += A*B with A=Ah+Al, B=Bh+Bl (drop Al*Bl)
__device__ __forceinline__ v8f wmma3(v16bf ah, v16bf al, v16bf bh, v16bf bl, v8f acc) {
  acc = __builtin_amdgcn_wmma_f32_16x16x32_bf16(false, ah, false, bl, (short)0, acc, false, false);
  acc = __builtin_amdgcn_wmma_f32_16x16x32_bf16(false, al, false, bh, (short)0, acc, false, false);
  acc = __builtin_amdgcn_wmma_f32_16x16x32_bf16(false, ah, false, bh, (short)0, acc, false, false);
  return acc;
}

// Packed fragment storage: fragment f, lane l -> 32 bf16 at (f*32+l)*32:
// [0..15] = hi, [16..31] = lo. 64B per lane, loads as 2x v16bf (4x b128).
__device__ __forceinline__ void load_frag(const __bf16* __restrict__ base,
                                          size_t f, int lane, v16bf& hi, v16bf& lo) {
  const v16bf* p = (const v16bf*)(base + ((size_t)f * 32 + lane) * 32);
  hi = p[0];
  lo = p[1];
}

// ---------------------------------------------------------------------------
// Pack B-layout fragments from row-major S[K x N].
// fragment f = kc*(N/16) + nt ; lane: col = nt*16 + (lane&15), K-half by lane>=16
// elem j -> S[kc*32 + a_kidx(j, half)][col]   (A-table K packing for B operands)
// ---------------------------------------------------------------------------
__global__ void pack_b_kernel(const float* __restrict__ S, __bf16* __restrict__ P,
                              int N, int nfrag) {
  int t = blockIdx.x * blockDim.x + threadIdx.x;
  int lane = t & 31, f = t >> 5;
  if (f >= nfrag) return;
  int nfr = N >> 4;
  int kc = f / nfr, nt = f % nfr;
  int half = lane >> 4, ln = lane & 15;
  const float* p = S + (size_t)(kc * 32) * N + nt * 16 + ln;
  __bf16* dst = P + ((size_t)f * 32 + lane) * 32;
#pragma unroll
  for (int j = 0; j < 16; ++j) {
    float x = p[(size_t)a_kidx(j, half) * N];
    __bf16 h = (__bf16)x;
    dst[j]      = h;
    dst[16 + j] = (__bf16)(x - (float)h);
  }
}

// ---------------------------------------------------------------------------
// Pack score-B fragments from roped K [KV x 512] (rows are keys -> contiguous).
// fragment f = h*(KV/16) + kt ; lane: key = kt*16 + (lane&15)
// elem j -> dim a_kidx(j, half) of that key's 32-dim head slice
// ---------------------------------------------------------------------------
__global__ void pack_kb_kernel(const float* __restrict__ Kr, __bf16* __restrict__ P) {
  int t = blockIdx.x * blockDim.x + threadIdx.x;
  int lane = t & 31, f = t >> 5;
  if (f >= HN * (KVLEN / 16)) return;
  int h = f / (KVLEN / 16), kt = f % (KVLEN / 16);
  int half = lane >> 4, ln = lane & 15;
  const float* p = Kr + (size_t)(kt * 16 + ln) * DKV + h * DH;
  __bf16* dst = P + ((size_t)f * 32 + lane) * 32;
#pragma unroll
  for (int j = 0; j < 16; ++j) {
    float x = p[a_kidx(j, half)];
    __bf16 h2 = (__bf16)x;
    dst[j]      = h2;
    dst[16 + j] = (__bf16)(x - (float)h2);
  }
}

// ---------------------------------------------------------------------------
// GEMM: C[M x N] = A[M x K] * Bpacked + bias[N].  8 waves/block, 16x64 per wave.
// Grid must be exactly (M/16)*(N/64)/8 blocks.
// ---------------------------------------------------------------------------
__global__ void __launch_bounds__(256) gemm_packedB_kernel(
    const float* __restrict__ A, const __bf16* __restrict__ Bp,
    const float* __restrict__ bias, float* __restrict__ C,
    int M, int N, int K) {
  const int wave = (blockIdx.x << 3) + (threadIdx.x >> 5);
  const int lane = threadIdx.x & 31;
  const int half = lane >> 4;
  const int ln   = lane & 15;
  const int ntn  = N >> 6;
  const int tm   = wave / ntn;
  const int tn   = wave % ntn;
  const int nfr  = N >> 4;

  const v8f zero = {0.f, 0.f, 0.f, 0.f, 0.f, 0.f, 0.f, 0.f};
  v8f acc[4] = {zero, zero, zero, zero};

  const float* arow = A + (size_t)(tm * 16 + ln) * K;

  for (int k0 = 0; k0 < K; k0 += 32) {
    float av[16];
    load16(arow + k0, half, av);
    v16bf ah, al;
    split16(av, 1.0f, ah, al);
    size_t f0 = (size_t)(k0 >> 5) * nfr + (tn << 2);
#pragma unroll
    for (int s = 0; s < 4; ++s) {
      v16bf bh, bl;
      load_frag(Bp, f0 + s, lane, bh, bl);
      acc[s] = wmma3(ah, al, bh, bl, acc[s]);
    }
  }

#pragma unroll
  for (int s = 0; s < 4; ++s) {
    const int col = (tn << 6) + s * 16 + ln;
    const float bv = bias[col];
#pragma unroll
    for (int r = 0; r < 8; ++r) {
      const int row = tm * 16 + r + half * 8;   // C/D layout: VGPR r -> row r (+8)
      C[(size_t)row * N + col] = acc[s][r] + bv;
    }
  }
}

// ---------------------------------------------------------------------------
__global__ void copy_f4_kernel(const f32x4* __restrict__ src,
                               f32x4* __restrict__ dst, int n4) {
  int i = blockIdx.x * blockDim.x + threadIdx.x;
  if (i < n4) dst[i] = src[i];
}

// In-place RoPE on [rows x 512] = [row][head*32 + d]; pos = row.
__global__ void rope_kernel(float* __restrict__ buf, int rows) {
  int tid = blockIdx.x * blockDim.x + threadIdx.x;
  if (tid >= rows * HN * 16) return;
  int i   = tid & 15;
  int h   = (tid >> 4) & 15;
  int row = tid >> 8;
  float inv_freq = __powf(10000.0f, -(float)i * (1.0f / 16.0f));
  float ang = (float)row * inv_freq;
  float c = __cosf(ang), s = __sinf(ang);
  size_t base = (size_t)row * DKV + h * DH + i;
  float x1 = buf[base];
  float x2 = buf[base + 16];
  buf[base]      = x1 * c - x2 * s;
  buf[base + 16] = x1 * s + x2 * c;
}

// ---------------------------------------------------------------------------
// Fused attention. One wave per (16-query tile, head).
// P1: scores via WMMA (packed K fragments) -> global + row maxima.
// P2: strip re-read (L2-resident, 256KB/block) -> both softmax denominators.
// P3: attn_weights out (non-temporal f32x4) + masked_w * V via WMMA (packed V).
// ---------------------------------------------------------------------------
__global__ void __launch_bounds__(32) mla_attn_kernel(
    const float* __restrict__ q_r,    // [LQ][512] roped
    const __bf16* __restrict__ pK,    // packed score-B fragments
    const __bf16* __restrict__ pV,    // packed V-B fragments
    const int*   __restrict__ vlens,  // [LQ]
    float* __restrict__ scores,       // [H][LQ][KV]
    float* __restrict__ attnw,        // [H][LQ][KV]
    float* __restrict__ heads) {      // [LQ][512]
  const int qt = blockIdx.x;
  const int h  = blockIdx.y;
  const int lane = threadIdx.x;
  const int half = lane >> 4;
  const int ln   = lane & 15;
  const float scale = 0.17677669529663687f;  // 1/sqrt(32), folded into q

  __shared__ float lds_mu[16], lds_su[16], lds_mm[16], lds_sm[16];
  const v8f zero = {0.f, 0.f, 0.f, 0.f, 0.f, 0.f, 0.f, 0.f};

  // Q tile A-fragment (row qt*16 + ln), scaled, built once per wave
  v16bf qh, qlo;
  {
    float qv[16];
    load16(q_r + (size_t)(qt * 16 + ln) * DKV + h * DH, half, qv);
    split16(qv, scale, qh, qlo);
  }

  float* sbase = scores + ((size_t)h * LQ + qt * 16) * KVLEN;
  const __bf16* kfb = pK + (size_t)h * (KVLEN / 16) * 1024;  // 32 lanes * 32 elems

  int   vl[8];
  float mm[8], mu[8];
#pragma unroll
  for (int r = 0; r < 8; ++r) {
    vl[r] = vlens[qt * 16 + r + half * 8];
    mm[r] = -10000.0f;   // reference mask value; valid_lens >= 1
    mu[r] = -3.0e38f;
  }

  // ---- P1: score tiles + row maxima ----
  for (int kv0 = 0; kv0 < KVLEN; kv0 += 16) {
    v16bf bh, bl;
    load_frag(kfb, (size_t)(kv0 >> 4), lane, bh, bl);
    v8f sc = wmma3(qh, qlo, bh, bl, zero);
    const int col = kv0 + ln;
#pragma unroll
    for (int r = 0; r < 8; ++r) {
      float s = sc[r];
      sbase[(size_t)(r + half * 8) * KVLEN + col] = s;
      mu[r] = fmaxf(mu[r], s);
      if (col < vl[r]) mm[r] = fmaxf(mm[r], s);
    }
  }
#pragma unroll
  for (int off = 8; off >= 1; off >>= 1) {
#pragma unroll
    for (int r = 0; r < 8; ++r) {
      mu[r] = fmaxf(mu[r], __shfl_xor(mu[r], off, 16));
      mm[r] = fmaxf(mm[r], __shfl_xor(mm[r], off, 16));
    }
  }

  // ---- P2: softmax denominators ----
  float su[8] = {0.f, 0.f, 0.f, 0.f, 0.f, 0.f, 0.f, 0.f};
  float sm[8] = {0.f, 0.f, 0.f, 0.f, 0.f, 0.f, 0.f, 0.f};
  for (int kv0 = 0; kv0 < KVLEN; kv0 += 16) {
    const int col = kv0 + ln;
#pragma unroll
    for (int r = 0; r < 8; ++r) {
      float s = sbase[(size_t)(r + half * 8) * KVLEN + col];
      su[r] += __expf(s - mu[r]);
      if (col < vl[r]) sm[r] += __expf(s - mm[r]);
    }
  }
#pragma unroll
  for (int off = 8; off >= 1; off >>= 1) {
#pragma unroll
    for (int r = 0; r < 8; ++r) {
      su[r] += __shfl_xor(su[r], off, 16);
      sm[r] += __shfl_xor(sm[r], off, 16);
    }
  }
  if (ln == 0) {
#pragma unroll
    for (int r = 0; r < 8; ++r) {
      lds_mu[r + half * 8] = mu[r];
      lds_su[r + half * 8] = su[r];
      lds_mm[r + half * 8] = mm[r];
      lds_sm[r + half * 8] = sm[r];
    }
  }
  __syncthreads();

  // ---- P3: attn_weights + masked_w * V ----
  const int   myrow  = qt * 16 + ln;
  const float mu_l   = lds_mu[ln];
  const float inv_su = 1.0f / lds_su[ln];
  const float mm_l   = lds_mm[ln];
  const float inv_sm = 1.0f / lds_sm[ln];
  const int   vl_l   = vlens[myrow];
  const float* srow = scores + ((size_t)h * LQ + myrow) * KVLEN;
  float*       arow = attnw  + ((size_t)h * LQ + myrow) * KVLEN;

  v8f o0 = zero, o1 = zero;
  for (int kv0 = 0; kv0 < KVLEN; kv0 += 32) {
    float sv[16];
    load16(srow + kv0, half, sv);            // A-layout positions, 4x b128
    float wu[16];
    v16bf wh, wl;
#pragma unroll
    for (int j = 0; j < 16; ++j) {
      const int kv = kv0 + a_kidx(j, half);
      wu[j] = __expf(sv[j] - mu_l) * inv_su;
      const float w = (kv < vl_l) ? __expf(sv[j] - mm_l) * inv_sm : 0.0f;
      __bf16 whj = (__bf16)w;
      wh[j] = whj;
      wl[j] = (__bf16)(w - (float)whj);
    }
    // non-temporal vector stores at the same A-layout positions
    f32x4* a4 = (f32x4*)(arow + kv0);
    f32x4 w0 = {wu[0],  wu[1],  wu[2],  wu[3]};
    f32x4 w1 = {wu[4],  wu[5],  wu[6],  wu[7]};
    f32x4 w2 = {wu[8],  wu[9],  wu[10], wu[11]};
    f32x4 w3 = {wu[12], wu[13], wu[14], wu[15]};
    __builtin_nontemporal_store(w0, a4 + half * 2 + 0);
    __builtin_nontemporal_store(w1, a4 + half * 2 + 1);
    __builtin_nontemporal_store(w2, a4 + half * 2 + 4);
    __builtin_nontemporal_store(w3, a4 + half * 2 + 5);

    // packed V fragments: f = (kv0/32)*(512/16) + h*2 + sub
    size_t fv = (size_t)(kv0 >> 5) * (DKV >> 4) + h * 2;
    v16bf bh, bl;
    load_frag(pV, fv + 0, lane, bh, bl);
    o0 = wmma3(wh, wl, bh, bl, o0);
    load_frag(pV, fv + 1, lane, bh, bl);
    o1 = wmma3(wh, wl, bh, bl, o1);
  }
#pragma unroll
  for (int r = 0; r < 8; ++r) {
    const size_t ro = (size_t)(qt * 16 + r + half * 8) * DKV + h * DH;
    heads[ro + ln]      = o0[r];
    heads[ro + 16 + ln] = o1[r];
  }
}

// ---------------------------------------------------------------------------
extern "C" void kernel_launch(void* const* d_in, const int* in_sizes, int n_in,
                              void* d_out, int out_size, void* d_ws, size_t ws_size,
                              hipStream_t stream) {
  (void)in_sizes; (void)n_in; (void)out_size; (void)ws_size;
  const float* x        = (const float*)d_in[0];
  const float* z_cache  = (const float*)d_in[1];
  const int*   vlens    = (const int*)  d_in[2];
  const float* W_latent = (const float*)d_in[3];
  const float* b_latent = (const float*)d_in[4];
  const float* W_q      = (const float*)d_in[5];
  const float* b_q      = (const float*)d_in[6];
  const float* W_k      = (const float*)d_in[7];
  const float* b_k      = (const float*)d_in[8];
  const float* W_v      = (const float*)d_in[9];
  const float* b_v      = (const float*)d_in[10];
  const float* W_o      = (const float*)d_in[11];
  const float* b_o      = (const float*)d_in[12];

  // d_out = [output | z | attn_weights | scores] (tuple order, flat)
  float* out    = (float*)d_out;
  float* z_out  = out   + (size_t)LQ * DMODEL;
  float* attnw  = z_out + (size_t)KVLEN * DKV;
  float* scores = attnw + (size_t)HN * LQ * KVLEN;

  // workspace (~54 MB)
  float* q_r    = (float*)d_ws;                 // 1,048,576 f
  float* k_r    = q_r + (size_t)LQ * DKV;       // 2,097,152 f
  float* v_l    = k_r + (size_t)KVLEN * DKV;    // 2,097,152 f
  float* heads  = v_l + (size_t)KVLEN * DKV;    // 1,048,576 f
  __bf16* pWlat = (__bf16*)(heads + (size_t)LQ * DKV);
  __bf16* pWq   = pWlat + (size_t)2048 * 1024;  // (2048/32)*(512/16) frags
  __bf16* pWk   = pWq   + (size_t)2048 * 1024;
  __bf16* pWv   = pWk   + (size_t)512 * 1024;   // (512/32)*(512/16) frags
  __bf16* pWo   = pWv   + (size_t)512 * 1024;
  __bf16* pK    = pWo   + (size_t)2048 * 1024;  // (512/32)*(2048/16) frags
  __bf16* pV    = pK    + (size_t)4096 * 1024;  // 16*(4096/16) frags

  const int NF_LAT = (DMODEL / 32) * (DKV / 16);    // 2048
  const int NF_KV  = (DKV / 32) * (DKV / 16);       // 512
  const int NF_O   = (DKV / 32) * (DMODEL / 16);    // 2048
  const int NF_V   = (KVLEN / 32) * (DKV / 16);     // 4096
  const int NF_K   = HN * (KVLEN / 16);             // 4096

  // 1) z = [z_cache ; x*W_latent + b_latent]
  {
    int n4 = (LQ * DKV) / 4;
    copy_f4_kernel<<<(n4 + 255) / 256, 256, 0, stream>>>(
        (const f32x4*)z_cache, (f32x4*)z_out, n4);
    pack_b_kernel<<<(NF_LAT * 32 + 255) / 256, 256, 0, stream>>>(W_latent, pWlat, DKV, NF_LAT);
    gemm_packedB_kernel<<<(LQ / 16) * (DKV / 64) / 8, 256, 0, stream>>>(
        x, pWlat, b_latent, z_out + (size_t)LQ * DKV, LQ, DKV, DMODEL);
  }
  // 2) q = rope(x*W_q + b_q)
  {
    pack_b_kernel<<<(NF_LAT * 32 + 255) / 256, 256, 0, stream>>>(W_q, pWq, DKV, NF_LAT);
    gemm_packedB_kernel<<<(LQ / 16) * (DKV / 64) / 8, 256, 0, stream>>>(
        x, pWq, b_q, q_r, LQ, DKV, DMODEL);
    rope_kernel<<<(LQ * HN * 16) / 256, 256, 0, stream>>>(q_r, LQ);
  }
  // 3) k = rope(z*W_k + b_k); pack score-B fragments
  {
    pack_b_kernel<<<(NF_KV * 32 + 255) / 256, 256, 0, stream>>>(W_k, pWk, DKV, NF_KV);
    gemm_packedB_kernel<<<(KVLEN / 16) * (DKV / 64) / 8, 256, 0, stream>>>(
        z_out, pWk, b_k, k_r, KVLEN, DKV, DKV);
    rope_kernel<<<(KVLEN * HN * 16) / 256, 256, 0, stream>>>(k_r, KVLEN);
    pack_kb_kernel<<<(NF_K * 32 + 255) / 256, 256, 0, stream>>>(k_r, pK);
  }
  // 4) v = z*W_v + b_v; pack V-B fragments
  {
    pack_b_kernel<<<(NF_KV * 32 + 255) / 256, 256, 0, stream>>>(W_v, pWv, DKV, NF_KV);
    gemm_packedB_kernel<<<(KVLEN / 16) * (DKV / 64) / 8, 256, 0, stream>>>(
        z_out, pWv, b_v, v_l, KVLEN, DKV, DKV);
    pack_b_kernel<<<(NF_V * 32 + 255) / 256, 256, 0, stream>>>(v_l, pV, DKV, NF_V);
  }
  // 5) fused attention: scores, attn_weights, heads
  {
    dim3 grid(LQ / 16, HN);
    mla_attn_kernel<<<grid, 32, 0, stream>>>(q_r, pK, pV, vlens, scores, attnw, heads);
  }
  // 6) output = heads*W_o + b_o
  {
    pack_b_kernel<<<(NF_O * 32 + 255) / 256, 256, 0, stream>>>(W_o, pWo, DMODEL, NF_O);
    gemm_packedB_kernel<<<(LQ / 16) * (DMODEL / 64) / 8, 256, 0, stream>>>(
        heads, pWo, b_o, out, LQ, DMODEL, DKV);
  }
}
